// TemporalRLSAttention_47038481826064
// MI455X (gfx1250) — matrix operand order repaired and checked by reading, not compiled, not run on hardware
//
#include <hip/hip_runtime.h>
#include <hip/hip_bf16.h>

#define DMODEL 128
#define BATCH 64
#define TSTEPS 4096
#define LAMBDA_REG 0.01f
#define FF 0.99f
#define INV_FF (1.0f/0.99f)
#define EPS 1e-8f
#define PROW (DMODEL + 1)    // f32 rows: 129 (odd stride, conflict-free)
#define XROW (DMODEL + 8)    // bf16 rows of length 128 -> 136 (272B, 16B-aligned)
#define BROW (BATCH + 8)     // bf16 rows of length 64  -> 72  (144B, 16B-aligned)

typedef __attribute__((ext_vector_type(16))) __bf16 v16bf;
typedef __attribute__((ext_vector_type(8)))  __bf16 v8bf;
typedef __attribute__((ext_vector_type(8)))  float  v8f;

static __device__ __forceinline__ v16bf join16(v8bf lo, v8bf hi) {
  return __builtin_shufflevector(lo, hi, 0, 1, 2, 3, 4, 5, 6, 7,
                                         8, 9, 10, 11, 12, 13, 14, 15);
}

// ---------------------------------------------------------------------------
// Phase 1: sequential RLS scan. One workgroup (8 wave32 waves) owns the whole
// recurrence. P (f32) lives in LDS; per-step GEMMs use v_wmma_f32_16x16x32_bf16
// with operands staged in LDS as bf16 tiles laid out so each fragment is two
// 16-byte ds_load_b128s. The x_t slice for step t+1 is loaded into registers
// while step t computes (software pipeline) so global latency never sits on
// the 4096-step critical path.
// ---------------------------------------------------------------------------
__global__ __launch_bounds__(256)
void rls_scan_kernel(const float* __restrict__ x,
                     const float* __restrict__ target,
                     float* __restrict__ attn_raw) {
  __shared__ float P[DMODEL][PROW];                         // 66 KB f32 master
  __shared__ float Pxs[BATCH][PROW];                        // 33 KB, Px (f32)
  __shared__ __attribute__((aligned(16))) __bf16 xA_bf[BATCH][XROW];   // x_t row-major
  __shared__ __attribute__((aligned(16))) __bf16 xT_bf[DMODEL][BROW];  // x_t^T
  __shared__ __attribute__((aligned(16))) __bf16 KT_bf[DMODEL][BROW];  // K^T
  __shared__ float theta_s[DMODEL];
  __shared__ float tgt_s[BATCH];
  __shared__ float coef_s[BATCH];                           // err*invden/B
  __shared__ float invden_s[BATCH];

  const int tid  = threadIdx.x;
  const int wave = tid >> 5;          // 0..7 (wave32)
  const int lane = tid & 31;
  const int half = lane >> 4;
  const int lm   = lane & 15;

  // this thread's slice chunk: fixed column group v, rows b0 + 8j (j=0..7)
  const int v  = (lane) * 4 & 127;    // (tid & 31) * 4
  const int b0 = tid >> 5;

  // init P = I / lambda, theta = 0
  for (int i = tid; i < DMODEL * DMODEL; i += 256) {
    int r = i >> 7, c = i & (DMODEL - 1);
    P[r][c] = (r == c) ? (1.0f / LAMBDA_REG) : 0.0f;
  }
  if (tid < DMODEL) theta_s[tid] = 0.0f;

  // preload slice t=0 into registers
  float4 xreg[8];
  #pragma unroll
  for (int j = 0; j < 8; ++j)
    xreg[j] = *(const float4*)(x + (size_t)(b0 + 8 * j) * TSTEPS * DMODEL + v);
  float tgt_reg = (tid < BATCH) ? target[(size_t)tid * TSTEPS] : 0.0f;
  __syncthreads();

  for (int t = 0; t < TSTEPS; ++t) {
    // ---- stage x_t from registers as bf16 (row-major + transposed) ----
    #pragma unroll
    for (int j = 0; j < 8; ++j) {
      const int b = b0 + 8 * j;
      const float4 f = xreg[j];
      xA_bf[b][v + 0] = (__bf16)f.x; xA_bf[b][v + 1] = (__bf16)f.y;
      xA_bf[b][v + 2] = (__bf16)f.z; xA_bf[b][v + 3] = (__bf16)f.w;
      xT_bf[v + 0][b] = (__bf16)f.x; xT_bf[v + 1][b] = (__bf16)f.y;
      xT_bf[v + 2][b] = (__bf16)f.z; xT_bf[v + 3][b] = (__bf16)f.w;
    }
    if (tid < BATCH) tgt_s[tid] = tgt_reg;
    // issue next slice's global loads now; they complete during this step
    if (t + 1 < TSTEPS) {
      #pragma unroll
      for (int j = 0; j < 8; ++j)
        xreg[j] = *(const float4*)(x + (size_t)(b0 + 8 * j) * TSTEPS * DMODEL
                                     + (size_t)(t + 1) * DMODEL + v);
      if (tid < BATCH) tgt_reg = target[(size_t)tid * TSTEPS + t + 1];
    }
    __syncthreads();

    // ---- GEMM 1: Px = x_t @ P^T   (M=64, N=128, K=128) ----
    // wave w owns N-tile n0 = 16*w. B (P) fragments built once per kk (each
    // P element read exactly once per wave); A fragments are two wide loads.
    {
      const int n0 = wave * 16;
      v8f acc[4] = {};
      #pragma unroll
      for (int kk = 0; kk < DMODEL; kk += 32) {
        v16bf bfrag;                           // B[k][n] = P[n][k]
        #pragma unroll
        for (int e = 0; e < 16; ++e)
          bfrag[e] = (__bf16)P[n0 + lm][kk + (e & 7) + ((e >> 3) << 4) + (half << 3)];
        #pragma unroll
        for (int m0 = 0; m0 < 4; ++m0) {
          const v8bf alo = *(const v8bf*)&xA_bf[m0 * 16 + lm][kk + (half << 3)];
          const v8bf ahi = *(const v8bf*)&xA_bf[m0 * 16 + lm][kk + 16 + (half << 3)];
          acc[m0] = __builtin_amdgcn_wmma_f32_16x16x32_bf16(
                      false, join16(alo, ahi), false, bfrag,
                      (short)0, acc[m0], false, false);
        }
      }
      #pragma unroll
      for (int m0 = 0; m0 < 4; ++m0)
        #pragma unroll
        for (int r = 0; r < 8; ++r)
          Pxs[m0 * 16 + r + half * 8][n0 + lm] = acc[m0][r];
    }
    __syncthreads();

    // ---- scalar RLS math (f32): denom, gain scale, y_pred, err ----
    if (tid < BATCH) {
      const int b = tid;
      float s = 0.0f, yp = 0.0f;
      #pragma unroll 4
      for (int d = 0; d < DMODEL; ++d) {
        const float xv = (float)xA_bf[b][d];
        s  += xv * Pxs[b][d];
        yp += xv * theta_s[d];
      }
      const float inv = 1.0f / (FF + s + EPS);
      invden_s[b] = inv;
      coef_s[b]   = (tgt_s[b] - yp) * inv * (1.0f / BATCH);
      attn_raw[(size_t)b * TSTEPS + t] = yp;                // logit
    }
    __syncthreads();

    // theta += mean_b(K * err); also stage K^T in bf16 for GEMM 2.
    if (tid < DMODEL) {
      const int d = tid;
      float acc = 0.0f;
      #pragma unroll 4
      for (int b = 0; b < BATCH; ++b) acc += Pxs[b][d] * coef_s[b];
      theta_s[d] += acc;
    }
    for (int i = tid; i < DMODEL * BATCH; i += 256) {
      const int b = i >> 7;           // consecutive threads -> consecutive m
      const int m = i & (DMODEL - 1);
      KT_bf[m][b] = (__bf16)(Pxs[b][m] * invden_s[b]);      // K[b][m]
    }
    __syncthreads();

    // ---- GEMM 2: dP = K^T @ x_t  (M=128, N=128, Kdim=64); P=(P-dP/B)/FF ----
    {
      const int m0 = wave * 16;
      v16bf afrag[2];
      #pragma unroll
      for (int bb2 = 0; bb2 < 2; ++bb2) {
        const v8bf alo = *(const v8bf*)&KT_bf[m0 + lm][bb2 * 32 + (half << 3)];
        const v8bf ahi = *(const v8bf*)&KT_bf[m0 + lm][bb2 * 32 + 16 + (half << 3)];
        afrag[bb2] = join16(alo, ahi);
      }
      v8f acc[8] = {};
      #pragma unroll
      for (int n0t = 0; n0t < 8; ++n0t) {
        #pragma unroll
        for (int bb2 = 0; bb2 < 2; ++bb2) {
          const v8bf blo = *(const v8bf*)&xT_bf[n0t * 16 + lm][bb2 * 32 + (half << 3)];
          const v8bf bhi = *(const v8bf*)&xT_bf[n0t * 16 + lm][bb2 * 32 + 16 + (half << 3)];
          acc[n0t] = __builtin_amdgcn_wmma_f32_16x16x32_bf16(
                       false, afrag[bb2], false, join16(blo, bhi),
                       (short)0, acc[n0t], false, false);
        }
      }
      #pragma unroll
      for (int n0t = 0; n0t < 8; ++n0t)
        #pragma unroll
        for (int r = 0; r < 8; ++r) {
          const int m = m0 + r + half * 8;
          const int n = n0t * 16 + lm;
          P[m][n] = (P[m][n] - acc[n0t][r] * (1.0f / BATCH)) * INV_FF;
        }
    }
    __syncthreads();
  }
}

// ---------------------------------------------------------------------------
// Phase 2: per-batch softmax over T (in place on the attn region of d_out) +
// output[b][d] = sum_t p[t] * x[b][t][d]. Bandwidth-bound (~128 MB read,
// ~5.5 us at 23.3 TB/s -> negligible vs the scan).
// ---------------------------------------------------------------------------
__global__ __launch_bounds__(256)
void softmax_einsum_kernel(const float* __restrict__ x,
                           float* __restrict__ out,
                           float* __restrict__ attn) {
  __shared__ float red[256];
  const int b   = blockIdx.x;
  const int tid = threadIdx.x;
  float* arow = attn + (size_t)b * TSTEPS;

  float m = -3.4e38f;
  for (int t = tid; t < TSTEPS; t += 256) m = fmaxf(m, arow[t]);
  red[tid] = m; __syncthreads();
  for (int s = 128; s > 0; s >>= 1) {
    if (tid < s) red[tid] = fmaxf(red[tid], red[tid + s]);
    __syncthreads();
  }
  m = red[0]; __syncthreads();

  float ssum = 0.0f;
  for (int t = tid; t < TSTEPS; t += 256) ssum += __expf(arow[t] - m);
  red[tid] = ssum; __syncthreads();
  for (int s = 128; s > 0; s >>= 1) {
    if (tid < s) red[tid] += red[tid + s];
    __syncthreads();
  }
  const float inv_s = 1.0f / red[0];
  __syncthreads();

  for (int t = tid; t < TSTEPS; t += 256)
    arow[t] = __expf(arow[t] - m) * inv_s;
  __syncthreads();

  const int d  = tid & 127;
  const int hh = tid >> 7;
  const float* xrow = x + (size_t)b * TSTEPS * DMODEL;
  float acc = 0.0f;
  for (int t = hh * (TSTEPS / 2); t < (hh + 1) * (TSTEPS / 2); ++t)
    acc = fmaf(arow[t], xrow[(size_t)t * DMODEL + d], acc);
  red[tid] = acc; __syncthreads();
  if (tid < DMODEL) out[(size_t)b * DMODEL + tid] = red[tid] + red[tid + 128];
}

// ---------------------------------------------------------------------------
extern "C" void kernel_launch(void* const* d_in, const int* in_sizes, int n_in,
                              void* d_out, int out_size, void* d_ws, size_t ws_size,
                              hipStream_t stream) {
  (void)in_sizes; (void)n_in; (void)out_size; (void)d_ws; (void)ws_size;
  const float* x      = (const float*)d_in[0];
  const float* target = (const float*)d_in[1];
  float* out  = (float*)d_out;                 // (B, D) = 8192 floats
  float* attn = out + BATCH * DMODEL;          // (B, T) = 262144 floats

  rls_scan_kernel<<<1, 256, 0, stream>>>(x, target, attn);
  softmax_einsum_kernel<<<BATCH, 256, 0, stream>>>(x, out, attn);
}
